// HybridLSTMContext_35862976921782
// MI455X (gfx1250) — compile-verified, matrix operand order checked by reading, not run on hardware
//
#include <hip/hip_runtime.h>
#include <hip/hip_bf16.h>

// ---------------- shapes ----------------
#define NC      151
#define EMBED   200
#define HID     512
#define G4      (4*HID)        // 2048
#define OBJ_DIM 4096
#define POS     128
#define OBJ_IN  (OBJ_DIM+EMBED+POS)   // 4424
#define DEC_IN  (OBJ_IN+HID)          // 4936
#define EDGE_IN (EMBED+HID+OBJ_DIM)   // 4808
#define NN      2048
#define BB      16
#define TT      (NN+BB)               // 2064
#define TT_P    2112                  // padded to multiple of 64 for 4x-M register blocking
// K padded to multiple of 32 for WMMA 16x16x32
#define OBJ_IN_P  4448
#define DEC_IN_P  4960
#define EDGE_IN_P 4832

typedef _Float16 v16h __attribute__((ext_vector_type(16)));
typedef _Float16 v8h  __attribute__((ext_vector_type(8)));
typedef float    v8f  __attribute__((ext_vector_type(8)));

__device__ __forceinline__ float sigf(float x) { return 1.0f / (1.0f + __expf(-x)); }

__device__ __forceinline__ v16h pack16(v8h lo, v8h hi) {
  v16h r;
#pragma unroll
  for (int i = 0; i < 8; ++i) { r[i] = lo[i]; r[i + 8] = hi[i]; }
  return r;
}

// ---------------- small prep kernels ----------------

// h1 = BN(box_info @ pos_W1 + b1), (NN x 32)
__global__ void k_pos_hidden(const float* __restrict__ box, const float* __restrict__ W1,
                             const float* __restrict__ b1, const float* __restrict__ g,
                             const float* __restrict__ be, const float* __restrict__ mu,
                             const float* __restrict__ var, float* __restrict__ h1) {
  int idx = blockIdx.x * blockDim.x + threadIdx.x;
  if (idx >= NN * 32) return;
  int r = idx >> 5, c = idx & 31;
  float s = b1[c];
#pragma unroll
  for (int k = 0; k < 9; ++k) s += box[r * 9 + k] * W1[k * 32 + c];
  s = (s - mu[c]) * rsqrtf(var[c] + 1e-5f) * g[c] + be[c];
  h1[idx] = s;
}

// feats f16, (TT_P x OBJ_IN_P), zero-padded K and rows
__global__ void k_build_feats(const float* __restrict__ x, const float* __restrict__ od,
                              const float* __restrict__ e1, const float* __restrict__ h1,
                              const float* __restrict__ W2, const float* __restrict__ b2,
                              const float* __restrict__ vobj, _Float16* __restrict__ feats) {
  int idx = blockIdx.x * blockDim.x + threadIdx.x;
  if (idx >= TT_P * OBJ_IN_P) return;
  int r = idx / OBJ_IN_P, c = idx % OBJ_IN_P;
  float v = 0.0f;
  if (c < OBJ_IN && r < TT) {
    if (r < NN) {
      if (c < OBJ_DIM) {
        v = x[(size_t)r * OBJ_DIM + c];
      } else if (c < OBJ_DIM + EMBED) {
        int e = c - OBJ_DIM;
        float s = 0.0f;
        for (int k = 0; k < NC; ++k) s += od[r * NC + k] * e1[k * EMBED + e];
        v = s;
      } else {
        int p = c - OBJ_DIM - EMBED;
        float s = b2[p];
#pragma unroll
        for (int k = 0; k < 32; ++k) s += h1[r * 32 + k] * W2[k * POS + p];
        v = fmaxf(s, 0.0f);
      }
    } else {
      v = vobj[(r - NN) * OBJ_IN + c];
    }
  }
  feats[idx] = (_Float16)v;
}

// WT[n*Kp + k] = (f16) W[k*G4 + n], zero pad k >= K
__global__ void k_transpose_wi(const float* __restrict__ W, _Float16* __restrict__ WT,
                               int K, int Kp) {
  int idx = blockIdx.x * blockDim.x + threadIdx.x;
  if (idx >= G4 * Kp) return;
  int n = idx / Kp, k = idx % Kp;
  WT[idx] = (k < K) ? (_Float16)W[(size_t)k * G4 + n] : (_Float16)0.0f;
}

// WhT[j*HID + k] = Wh[k*G4 + j]   (f32, for the recurrent matvec)
__global__ void k_transpose_wh(const float* __restrict__ Wh, float* __restrict__ WhT) {
  int idx = blockIdx.x * blockDim.x + threadIdx.x;
  if (idx >= G4 * HID) return;
  int j = idx / HID, k = idx % HID;
  WhT[idx] = Wh[(size_t)k * G4 + j];
}

// dec input f16: [feats | enc_h | pad], (TT_P x DEC_IN_P)
__global__ void k_build_decinp(const _Float16* __restrict__ feats, const float* __restrict__ eh,
                               _Float16* __restrict__ di) {
  int idx = blockIdx.x * blockDim.x + threadIdx.x;
  if (idx >= TT_P * DEC_IN_P) return;
  int r = idx / DEC_IN_P, c = idx % DEC_IN_P;
  _Float16 v = (_Float16)0.0f;
  if (r < TT) {
    if (c < OBJ_IN)      v = feats[(size_t)r * OBJ_IN_P + c];
    else if (c < DEC_IN) v = (_Float16)eh[(size_t)r * HID + (c - OBJ_IN)];
  }
  di[idx] = v;
}

// edge feats f16: [embed2[pred] | x | enc_h | pad] / virtual_edge, (TT_P x EDGE_IN_P)
__global__ void k_build_edgefeats(const float* __restrict__ e2, const int* __restrict__ pred,
                                  const float* __restrict__ x, const float* __restrict__ eh,
                                  const float* __restrict__ vedge, _Float16* __restrict__ ef) {
  int idx = blockIdx.x * blockDim.x + threadIdx.x;
  if (idx >= TT_P * EDGE_IN_P) return;
  int r = idx / EDGE_IN_P, c = idx % EDGE_IN_P;
  float v = 0.0f;
  if (c < EDGE_IN && r < TT) {
    if (r < NN) {
      if (c < EMBED)                 v = e2[pred[r] * EMBED + c];
      else if (c < EMBED + OBJ_DIM)  v = x[(size_t)r * OBJ_DIM + (c - EMBED)];
      else                           v = eh[(size_t)r * HID + (c - EMBED - OBJ_DIM)];
    } else {
      v = vedge[(r - NN) * EDGE_IN + c];
    }
  }
  ef[idx] = (_Float16)v;
}

// ---------------- register-blocked WMMA GEMM ----------------
// C[Mp x 2048] = A[Mp x Kp](f16) * Wi + bias, with BT = Wi^T rows (2048 x Kp, f16).
// Each wave computes a 64x32 macro-tile: MT=4 M-tiles x NT=2 N-tiles of 16x16,
// reusing each A fragment across NT WMMAs and each B fragment across MT WMMAs.
// Per K=32 step: 6 fragment loads (192 B/lane) feed 8 v_wmma_f32_16x16x32_f16.
// Mp and grid sizing guarantee full tiles -> EXEC stays all-ones (WMMA requirement).
#define GMT 4
#define GNT 2
__global__ void __launch_bounds__(128)
k_wmma_gemm_bias(const _Float16* __restrict__ A, const _Float16* __restrict__ BT,
                 const float* __restrict__ bias, float* __restrict__ C,
                 int Kp) {
  const int lane = threadIdx.x & 31;
  const int wave = threadIdx.x >> 5;               // 4 waves: split along N
  const int m0   = blockIdx.y * (GMT * 16);        // 64-row strip
  const int n0   = blockIdx.x * (4 * GNT * 16) + wave * (GNT * 16);
  const int half = lane >> 4;
  const int l15  = lane & 15;

  const _Float16* arow[GMT];
  const _Float16* brow[GNT];
#pragma unroll
  for (int i = 0; i < GMT; ++i) arow[i] = A  + (size_t)(m0 + 16 * i + l15) * Kp + 8 * half;
#pragma unroll
  for (int j = 0; j < GNT; ++j) brow[j] = BT + (size_t)(n0 + 16 * j + l15) * Kp + 8 * half;

  v8f acc[GMT][GNT];
#pragma unroll
  for (int i = 0; i < GMT; ++i)
#pragma unroll
    for (int j = 0; j < GNT; ++j) acc[i][j] = (v8f){};

  for (int k0 = 0; k0 < Kp; k0 += 32) {
    v16h a[GMT], b[GNT];
#pragma unroll
    for (int i = 0; i < GMT; ++i) {
      v8h lo = *(const v8h*)(arow[i] + k0);
      v8h hi = *(const v8h*)(arow[i] + k0 + 16);
      a[i] = pack16(lo, hi);
    }
#pragma unroll
    for (int j = 0; j < GNT; ++j) {
      v8h lo = *(const v8h*)(brow[j] + k0);
      v8h hi = *(const v8h*)(brow[j] + k0 + 16);
      b[j] = pack16(lo, hi);
    }
    __builtin_prefetch(arow[0] + k0 + 128, 0, 1);  // global_prefetch_b8
    __builtin_prefetch(brow[0] + k0 + 128, 0, 1);
#pragma unroll
    for (int i = 0; i < GMT; ++i)
#pragma unroll
      for (int j = 0; j < GNT; ++j)
        acc[i][j] = __builtin_amdgcn_wmma_f32_16x16x32_f16(
            false, a[i], false, b[j], (short)0, acc[i][j], false, false);
  }

#pragma unroll
  for (int j = 0; j < GNT; ++j) {
    const int col = n0 + 16 * j + l15;
    const float bv = bias[col];
#pragma unroll
    for (int i = 0; i < GMT; ++i) {
#pragma unroll
      for (int v = 0; v < 8; ++v) {                // C layout: VGPR v -> M = v + 8*half
        int row = m0 + 16 * i + v + 8 * half;
        C[(size_t)row * G4 + col] = acc[i][j][v] + bv;
      }
    }
  }
}

// ---------------- sequential LSTM scan (one persistent 1024-thread block) ----------------
__global__ void __launch_bounds__(1024)
k_lstm_scan(const float* __restrict__ pre, const float* __restrict__ WhT,
            float* __restrict__ hout) {
  __shared__ float sh[HID];
  __shared__ float sc[HID];
  __shared__ float sg[G4];
  const int tid = threadIdx.x;
  if (tid < HID) { sh[tid] = 0.0f; sc[tid] = 0.0f; }
  __syncthreads();
  for (int t = 0; t < TT; ++t) {
    const float* pt = pre + (size_t)t * G4;
    for (int j = tid; j < G4; j += 1024) {
      const float* w = WhT + (size_t)j * HID;
      float s = pt[j];
#pragma unroll 8
      for (int k = 0; k < HID; ++k) s += sh[k] * w[k];
      sg[j] = s;
    }
    __syncthreads();
    if (tid < HID) {
      float gi = sg[tid], gf = sg[HID + tid], gg = sg[2 * HID + tid], go = sg[3 * HID + tid];
      float c = sigf(gf) * sc[tid] + sigf(gi) * tanhf(gg);
      float h = sigf(go) * tanhf(c);
      sc[tid] = c; sh[tid] = h;
      hout[(size_t)t * HID + tid] = h;
    }
    __syncthreads();
  }
}

// ---------------- output projection + argmax + copy ----------------
__global__ void k_out_proj(const float* __restrict__ dh, const float* __restrict__ W,
                           const float* __restrict__ b, float* __restrict__ dists) {
  int idx = blockIdx.x * blockDim.x + threadIdx.x;
  if (idx >= NN * NC) return;
  int r = idx / NC, c = idx % NC;
  float s = b[c];
  const float* h = dh + (size_t)r * HID;
#pragma unroll 8
  for (int k = 0; k < HID; ++k) s += h[k] * W[k * NC + c];
  dists[idx] = s;
}

__global__ void k_argmax(const float* __restrict__ dists, float* __restrict__ pred_f,
                         int* __restrict__ pred_i) {
  int r = blockIdx.x * blockDim.x + threadIdx.x;
  if (r >= NN) return;
  const float* d = dists + (size_t)r * NC;
  int best = 0; float bv = d[0];
  for (int c = 1; c < NC; ++c) { float v = d[c]; if (v > bv) { bv = v; best = c; } }
  pred_i[r] = best;
  pred_f[r] = (float)best;
}

__global__ void k_copy_edge(const float* __restrict__ eh, float* __restrict__ out) {
  int idx = blockIdx.x * blockDim.x + threadIdx.x;
  if (idx >= NN * HID) return;
  out[idx] = eh[idx];
}

// ---------------- launch ----------------
extern "C" void kernel_launch(void* const* d_in, const int* in_sizes, int n_in,
                              void* d_out, int out_size, void* d_ws, size_t ws_size,
                              hipStream_t stream) {
  const float* x      = (const float*)d_in[0];
  const float* od     = (const float*)d_in[1];
  const float* box    = (const float*)d_in[2];
  const float* vobj   = (const float*)d_in[3];
  const float* vedge  = (const float*)d_in[4];
  const float* e1     = (const float*)d_in[5];
  const float* e2     = (const float*)d_in[6];
  const float* pW1    = (const float*)d_in[7];
  const float* pb1    = (const float*)d_in[8];
  const float* bng    = (const float*)d_in[9];
  const float* bnb    = (const float*)d_in[10];
  const float* bnm    = (const float*)d_in[11];
  const float* bnv    = (const float*)d_in[12];
  const float* pW2    = (const float*)d_in[13];
  const float* pb2    = (const float*)d_in[14];
  const float* objWi  = (const float*)d_in[15];
  const float* objWh  = (const float*)d_in[16];
  const float* objb   = (const float*)d_in[17];
  const float* decWi  = (const float*)d_in[18];
  const float* decWh  = (const float*)d_in[19];
  const float* decb   = (const float*)d_in[20];
  const float* outW   = (const float*)d_in[21];
  const float* outb   = (const float*)d_in[22];
  const float* edgWi  = (const float*)d_in[23];
  const float* edgWh  = (const float*)d_in[24];
  const float* edgb   = (const float*)d_in[25];

  float* out_dists = (float*)d_out;                       // 2048*151
  float* out_pred  = out_dists + (size_t)NN * NC;         // 2048
  float* out_edge  = out_pred + NN;                       // 2048*512

  // workspace carve-up (256B aligned regions)
  char* w = (char*)d_ws;
  auto alloc = [&](size_t bytes) -> char* {
    char* p = w; w += (bytes + 255) & ~(size_t)255; return p;
  };
  _Float16* feats   = (_Float16*)alloc((size_t)TT_P * OBJ_IN_P * 2);
  _Float16* decinp  = (_Float16*)alloc((size_t)TT_P * DEC_IN_P * 2);
  _Float16* edgef   = (_Float16*)alloc((size_t)TT_P * EDGE_IN_P * 2);
  _Float16* objWiT  = (_Float16*)alloc((size_t)G4 * OBJ_IN_P * 2);
  _Float16* decWiT  = (_Float16*)alloc((size_t)G4 * DEC_IN_P * 2);
  _Float16* edgWiT  = (_Float16*)alloc((size_t)G4 * EDGE_IN_P * 2);
  float*    objWhT  = (float*)alloc((size_t)G4 * HID * 4);
  float*    decWhT  = (float*)alloc((size_t)G4 * HID * 4);
  float*    edgWhT  = (float*)alloc((size_t)G4 * HID * 4);
  float*    pre     = (float*)alloc((size_t)TT_P * G4 * 4);   // reused by all 3 LSTMs
  float*    enc_h   = (float*)alloc((size_t)TT * HID * 4);
  float*    dec_h   = (float*)alloc((size_t)TT * HID * 4);
  float*    edg_h   = (float*)alloc((size_t)TT * HID * 4);
  float*    h1      = (float*)alloc((size_t)NN * 32 * 4);
  int*      pred_i  = (int*)alloc((size_t)NN * 4);

  const int TPB = 256;
  auto nb = [](size_t n, int tpb) { return (int)((n + tpb - 1) / tpb); };

  // ---- stage 1: embeddings + feats assembly ----
  k_pos_hidden<<<nb((size_t)NN * 32, TPB), TPB, 0, stream>>>(box, pW1, pb1, bng, bnb, bnm, bnv, h1);
  k_build_feats<<<nb((size_t)TT_P * OBJ_IN_P, TPB), TPB, 0, stream>>>(x, od, e1, h1, pW2, pb2, vobj, feats);

  // ---- weight reshapes (f16 transposed Wi; f32 transposed Wh) ----
  k_transpose_wi<<<nb((size_t)G4 * OBJ_IN_P, TPB), TPB, 0, stream>>>(objWi, objWiT, OBJ_IN, OBJ_IN_P);
  k_transpose_wi<<<nb((size_t)G4 * DEC_IN_P, TPB), TPB, 0, stream>>>(decWi, decWiT, DEC_IN, DEC_IN_P);
  k_transpose_wi<<<nb((size_t)G4 * EDGE_IN_P, TPB), TPB, 0, stream>>>(edgWi, edgWiT, EDGE_IN, EDGE_IN_P);
  k_transpose_wh<<<nb((size_t)G4 * HID, TPB), TPB, 0, stream>>>(objWh, objWhT);
  k_transpose_wh<<<nb((size_t)G4 * HID, TPB), TPB, 0, stream>>>(decWh, decWhT);
  k_transpose_wh<<<nb((size_t)G4 * HID, TPB), TPB, 0, stream>>>(edgWh, edgWhT);

  // grid: x covers N in 128-col blocks (4 waves x 32 cols), y covers M in 64-row strips
  dim3 gemm_grid(G4 / 128, TT_P / 64);   // 16 x 33

  // ---- encoder LSTM ----
  k_wmma_gemm_bias<<<gemm_grid, 128, 0, stream>>>(feats, objWiT, objb, pre, OBJ_IN_P);
  k_lstm_scan<<<1, 1024, 0, stream>>>(pre, objWhT, enc_h);

  // ---- decoder LSTM ----
  k_build_decinp<<<nb((size_t)TT_P * DEC_IN_P, TPB), TPB, 0, stream>>>(feats, enc_h, decinp);
  k_wmma_gemm_bias<<<gemm_grid, 128, 0, stream>>>(decinp, decWiT, decb, pre, DEC_IN_P);
  k_lstm_scan<<<1, 1024, 0, stream>>>(pre, decWhT, dec_h);

  // ---- classification head ----
  k_out_proj<<<nb((size_t)NN * NC, TPB), TPB, 0, stream>>>(dec_h, outW, outb, out_dists);
  k_argmax<<<nb((size_t)NN, TPB), TPB, 0, stream>>>(out_dists, out_pred, pred_i);

  // ---- edge LSTM ----
  k_build_edgefeats<<<nb((size_t)TT_P * EDGE_IN_P, TPB), TPB, 0, stream>>>(e2, pred_i, x, enc_h, vedge, edgef);
  k_wmma_gemm_bias<<<gemm_grid, 128, 0, stream>>>(edgef, edgWiT, edgb, pre, EDGE_IN_P);
  k_lstm_scan<<<1, 1024, 0, stream>>>(pre, edgWhT, edg_h);
  k_copy_edge<<<nb((size_t)NN * HID, TPB), TPB, 0, stream>>>(edg_h, out_edge);

  (void)in_sizes; (void)n_in; (void)out_size; (void)ws_size;
}
